// CrystalGraphConvNet_pyggit_41875931136507
// MI455X (gfx1250) — matrix operand order repaired and checked by reading, not compiled, not run on hardware
//
#include <hip/hip_runtime.h>
#include <hip/hip_bf16.h>
#include <math.h>

// ---------------- problem constants (CGCNN) ----------------
#define NNODES 50000
#define NEDGES 800000
#define NGRAPH 512
#define F0     92
#define FDIM   64      // atom feature len
#define NB     41      // edge feature len
#define HDIM   128
#define NCONV  3
#define KPAD   192     // 2F + NB = 169, padded to 6*32 for bf16 WMMA
#define KT     6       // k-steps of 32
#define NT     4       // 16-col n-tiles covering 64
#define EPSBN  1e-5f

typedef __attribute__((ext_vector_type(16))) __bf16 v16bf;
typedef __attribute__((ext_vector_type(8)))  float  v8f;

// fragments per layer: 2 mats * 6 kt * 4 nt, each 32 lanes * 16 bf16
#define FRAGS_PER_LAYER (2 * KT * NT)
#define PACK_PER_LAYER  (FRAGS_PER_LAYER * 32 * 16)   // bf16 elements

// ---- fast transcendentals: v_exp_f32 / v_log_f32 / v_rcp_f32 (TRANS pipe,
// co-executes with VALU on CDNA5) ----
__device__ __forceinline__ float fexp(float x)  { return __expf(x); }
__device__ __forceinline__ float flog(float x)  { return __logf(x); }
__device__ __forceinline__ float frcp(float x)  { return __builtin_amdgcn_rcpf(x); }

// stable softplus: log argument stays in [1,2] -> no cancellation
__device__ __forceinline__ float softplusf(float x) {
    return fmaxf(x, 0.f) + flog(1.f + fexp(-fabsf(x)));
}
__device__ __forceinline__ float sigmoidf_(float x) {
    return frcp(1.f + fexp(-x));
}
__device__ __forceinline__ void atomAddF(float* p, float v) {
    __hip_atomic_fetch_add(p, v, __ATOMIC_RELAXED, __HIP_MEMORY_SCOPE_AGENT);
}

// ---------------- generic zero ----------------
__global__ void zero_f32(float* __restrict__ p, long n) {
    long i = (long)blockIdx.x * blockDim.x + threadIdx.x;
    if (i < n) p[i] = 0.f;
}

// ---------------- pack weights into WMMA B-fragment layout ----------------
// packed[layer][mat][kt][nt][lane][j], lane: N = nt*16 + (lane&15),
// K = kt*32 + (lane>>4)*16 + j   (zero-padded past K=169)
__global__ void pack_weights(const float* __restrict__ lnf_W,
                             const float* __restrict__ lns_W,
                             __bf16* __restrict__ packed) {
    int idx = blockIdx.x * blockDim.x + threadIdx.x;   // (layer,mat,kt,nt,lane)
    const int total = NCONV * 2 * KT * NT * 32;
    if (idx >= total) return;
    int lane = idx & 31;
    int t = idx >> 5;
    int nt = t & 3;  t >>= 2;
    int kt = t % KT; t /= KT;
    int mat = t & 1; int layer = t >> 1;
    const float* W = (mat == 0 ? lnf_W : lns_W) + (size_t)layer * (2 * FDIM + NB) * FDIM;
    int n = nt * 16 + (lane & 15);
    __bf16* outp = packed + (size_t)idx * 16;
    #pragma unroll
    for (int j = 0; j < 16; ++j) {
        int k = kt * 32 + (lane >> 4) * 16 + j;
        float v = (k < 2 * FDIM + NB) ? W[(size_t)k * FDIM + n] : 0.f;
        outp[j] = (__bf16)v;
    }
}

// ---------------- embedding: h = x @ emb_W + emb_b ----------------
__global__ void embed_kernel(const float* __restrict__ x,
                             const float* __restrict__ W,
                             const float* __restrict__ b,
                             float* __restrict__ h) {
    __shared__ float xr[F0];
    int n = blockIdx.x;
    int f = threadIdx.x;                  // 64 threads
    for (int k = f; k < F0; k += FDIM) xr[k] = x[(size_t)n * F0 + k];
    __syncthreads();
    float acc = b[f];
    #pragma unroll 4
    for (int k = 0; k < F0; ++k) acc = fmaf(xr[k], W[(size_t)k * FDIM + f], acc);
    h[(size_t)n * FDIM + f] = acc;
}

// ---------------- BN stats: per-feature sum / sumsq over nodes ----------------
__global__ void bn_stats(const float* __restrict__ h, float* __restrict__ sum,
                         float* __restrict__ sumsq, int n) {
    int f = threadIdx.x & 63;
    int grp = threadIdx.x >> 6;           // 4 row streams per block
    float s = 0.f, ss = 0.f;
    for (long r = (long)blockIdx.x * 4 + grp; r < n; r += (long)gridDim.x * 4) {
        float v = h[r * FDIM + f];
        s += v; ss += v * v;
    }
    __shared__ float sh[256], sh2[256];
    sh[threadIdx.x] = s; sh2[threadIdx.x] = ss;
    __syncthreads();
    if (grp == 0) {
        s  = sh[f]  + sh[64 + f]  + sh[128 + f]  + sh[192 + f];
        ss = sh2[f] + sh2[64 + f] + sh2[128 + f] + sh2[192 + f];
        atomAddF(&sum[f], s);
        atomAddF(&sumsq[f], ss);
    }
}

// ---------------- BN1 apply + f32/bf16 dual write, zero agg ----------------
__global__ void bn_apply_cast(const float* __restrict__ h,
                              const float* __restrict__ sum,
                              const float* __restrict__ sumsq,
                              const float* __restrict__ g,
                              const float* __restrict__ b, int n,
                              float* __restrict__ xn, __bf16* __restrict__ xnb,
                              float* __restrict__ agg) {
    long i = (long)blockIdx.x * blockDim.x + threadIdx.x;
    long tot = (long)n * FDIM;
    if (i >= tot) return;
    int f = (int)(i & 63);
    float invn = 1.f / (float)n;
    float mean = sum[f] * invn;
    float var  = sumsq[f] * invn - mean * mean;
    float v = (h[i] - mean) * rsqrtf(var + EPSBN) * g[f] + b[f];
    xn[i] = v;
    xnb[i] = (__bf16)v;
    agg[i] = 0.f;
}

// ---------------- BN2 apply + residual + softplus ----------------
__global__ void bn2_residual(const float* __restrict__ agg,
                             const float* __restrict__ sum,
                             const float* __restrict__ sumsq,
                             const float* __restrict__ g,
                             const float* __restrict__ b, int n,
                             const float* __restrict__ xn, float* __restrict__ hout) {
    long i = (long)blockIdx.x * blockDim.x + threadIdx.x;
    long tot = (long)n * FDIM;
    if (i >= tot) return;
    int f = (int)(i & 63);
    float invn = 1.f / (float)n;
    float mean = sum[f] * invn;
    float var  = sumsq[f] * invn - mean * mean;
    float v = (agg[i] - mean) * rsqrtf(var + EPSBN) * g[f] + b[f] + xn[i];
    hout[i] = softplusf(v);
}

// ---------------- THE hot kernel: fused gather + dual WMMA GEMM + gate + scatter ----
// One wave handles a tile of 16 edges, producing the 16x64 gated message and
// atomically scattering it to agg[src]. A fragments are assembled directly from
// global gathers (bf16 node rows + on-the-fly converted edge_attr); B fragments
// come from LDS (pre-packed). 48 v_wmma_f32_16x16x32_bf16 per tile.
union AFrag { v16bf v; uint4 q[2]; };

__device__ __forceinline__ uint4 load_chunk(int c,
                                            const __bf16* __restrict__ rowS,
                                            const __bf16* __restrict__ rowD,
                                            const float* __restrict__ ea, long e) {
    if (c < 8)  return *(const uint4*)(rowS + c * 8);
    if (c < 16) return *(const uint4*)(rowD + (c - 8) * 8);
    if (c < 22) {
        union { uint4 q; __bf16 h[8]; } u;
        int base = (c - 16) * 8;
        #pragma unroll
        for (int j = 0; j < 8; ++j) {
            int col = base + j;
            float v = (col < NB) ? ea[e * NB + col] : 0.f;
            u.h[j] = (__bf16)v;
        }
        return u.q;
    }
    uint4 z; z.x = 0u; z.y = 0u; z.z = 0u; z.w = 0u;
    return z;
}

__global__ __launch_bounds__(256) void edge_conv_wmma(
    const __bf16* __restrict__ xnb,
    const int* __restrict__ src, const int* __restrict__ dst,
    const float* __restrict__ ea,
    const __bf16* __restrict__ packedL,   // this layer's fragments
    const float* __restrict__ bfv, const float* __restrict__ bsv,
    float* __restrict__ agg, int ntiles) {

    __shared__ v16bf Bsh[FRAGS_PER_LAYER * 32];       // 48 KB
    {   // cooperative LDS fill of packed B fragments
        const uint4* gp = (const uint4*)packedL;
        uint4* lp = (uint4*)Bsh;
        const int n16 = FRAGS_PER_LAYER * 32 * 2;     // 16B chunks
        for (int i = threadIdx.x; i < n16; i += blockDim.x) lp[i] = gp[i];
    }
    __syncthreads();

    const int lane = threadIdx.x & 31;
    const int hi = lane >> 4;             // K-half group
    const int m  = lane & 15;             // edge row / output column

    float bF[NT], bS[NT];
    #pragma unroll
    for (int nt = 0; nt < NT; ++nt) { bF[nt] = bfv[nt * 16 + m]; bS[nt] = bsv[nt * 16 + m]; }

    const v8f vzero = {0.f, 0.f, 0.f, 0.f, 0.f, 0.f, 0.f, 0.f};

    int wave = blockIdx.x * (blockDim.x >> 5) + (threadIdx.x >> 5);
    int wstride = gridDim.x * (blockDim.x >> 5);

    for (int tile = wave; tile < ntiles; tile += wstride) {
        const long e0 = (long)tile * 16;
        const long e  = e0 + m;
        const long s  = src[e];
        const long d  = dst[e];
        const __bf16* rowS = xnb + s * FDIM;
        const __bf16* rowD = xnb + d * FDIM;

        v8f accF[NT], accS[NT];
        #pragma unroll
        for (int nt = 0; nt < NT; ++nt) { accF[nt] = vzero; accS[nt] = vzero; }

        #pragma unroll
        for (int kt = 0; kt < KT; ++kt) {
            AFrag a;
            a.q[0] = load_chunk(4 * kt + 2 * hi,     rowS, rowD, ea, e);
            a.q[1] = load_chunk(4 * kt + 2 * hi + 1, rowS, rowD, ea, e);
            #pragma unroll
            for (int nt = 0; nt < NT; ++nt) {
                v16bf bWf = Bsh[((0 * KT + kt) * NT + nt) * 32 + lane];
                accF[nt] = __builtin_amdgcn_wmma_f32_16x16x32_bf16(
                    false, a.v, false, bWf, (short)0, accF[nt], false, false);
                v16bf bWs = Bsh[((1 * KT + kt) * NT + nt) * 32 + lane];
                accS[nt] = __builtin_amdgcn_wmma_f32_16x16x32_bf16(
                    false, a.v, false, bWs, (short)0, accS[nt], false, false);
            }
        }

        // epilogue: m = sigmoid(zWf+bf) * softplus(zWs+bs); scatter to agg[src]
        int sidx[8];
        #pragma unroll
        for (int r = 0; r < 8; ++r) sidx[r] = src[e0 + hi * 8 + r];

        #pragma unroll
        for (int nt = 0; nt < NT; ++nt) {
            const int N = nt * 16 + m;
            #pragma unroll
            for (int r = 0; r < 8; ++r) {
                float fv = accF[nt][r] + bF[nt];
                float sv = accS[nt][r] + bS[nt];
                float msg = sigmoidf_(fv) * softplusf(sv);
                atomAddF(&agg[(long)sidx[r] * FDIM + N], msg);
            }
        }
    }
}

// ---------------- global mean pool (scatter) ----------------
__global__ void pool_kernel(const float* __restrict__ h, const int* __restrict__ batch,
                            float* __restrict__ sums, float* __restrict__ cnt, int n) {
    long i = (long)blockIdx.x * blockDim.x + threadIdx.x;
    if (i >= (long)n * FDIM) return;
    int node = (int)(i >> 6);
    int f = (int)(i & 63);
    int gidx = batch[node];
    atomAddF(&sums[(long)gidx * FDIM + f], h[i]);
    if (f == 0) atomAddF(&cnt[gidx], 1.f);
}

// ---------------- readout MLP: softplus->fc1->softplus->out ----------------
__global__ void readout_kernel(const float* __restrict__ sums, const float* __restrict__ cnt,
                               const float* __restrict__ W1, const float* __restrict__ b1,
                               const float* __restrict__ W2, const float* __restrict__ b2,
                               float* __restrict__ out) {
    int g = blockIdx.x;
    int t = threadIdx.x;                  // 128
    __shared__ float sc[FDIM];
    __shared__ float red[HDIM];
    float c = fmaxf(cnt[g], 1.f);
    if (t < FDIM) sc[t] = softplusf(sums[(long)g * FDIM + t] / c);
    __syncthreads();
    float a = b1[t];
    #pragma unroll 8
    for (int f = 0; f < FDIM; ++f) a = fmaf(sc[f], W1[(size_t)f * HDIM + t], a);
    red[t] = softplusf(a) * W2[t];
    __syncthreads();
    for (int s2 = HDIM / 2; s2 > 0; s2 >>= 1) {
        if (t < s2) red[t] += red[t + s2];
        __syncthreads();
    }
    if (t == 0) out[g] = red[0] + b2[0];
}

// ---------------- host orchestration ----------------
extern "C" void kernel_launch(void* const* d_in, const int* in_sizes, int n_in,
                              void* d_out, int out_size, void* d_ws, size_t ws_size,
                              hipStream_t stream) {
    const float* x        = (const float*)d_in[0];
    const int*   ei       = (const int*)d_in[1];
    const float* ea       = (const float*)d_in[2];
    const int*   batch    = (const int*)d_in[3];
    const float* emb_W    = (const float*)d_in[4];
    const float* emb_b    = (const float*)d_in[5];
    const float* bn1_g    = (const float*)d_in[6];
    const float* bn1_b    = (const float*)d_in[7];
    const float* bn2_g    = (const float*)d_in[8];
    const float* bn2_b    = (const float*)d_in[9];
    const float* lnf_W    = (const float*)d_in[10];
    const float* lnf_b    = (const float*)d_in[11];
    const float* lns_W    = (const float*)d_in[12];
    const float* lns_b    = (const float*)d_in[13];
    const float* fc1_W    = (const float*)d_in[14];
    const float* fc1_b    = (const float*)d_in[15];
    const float* out_W    = (const float*)d_in[16];
    const float* out_b    = (const float*)d_in[17];
    float* out = (float*)d_out;

    const int N = in_sizes[0] / F0;       // 50000
    const int E = in_sizes[1] / 2;        // 800000
    const int G = out_size;               // 512
    const int ntiles = E / 16;

    // carve workspace (256B aligned blocks)
    char* w = (char*)d_ws;
    auto carve = [&](size_t bytes) -> void* {
        void* p = (void*)w;
        w += (bytes + 255) & ~(size_t)255;
        return p;
    };
    float*  h      = (float*) carve((size_t)N * FDIM * 4);
    float*  xn     = (float*) carve((size_t)N * FDIM * 4);
    __bf16* xnb    = (__bf16*)carve((size_t)N * FDIM * 2);
    float*  agg    = (float*) carve((size_t)N * FDIM * 4);
    __bf16* packed = (__bf16*)carve((size_t)NCONV * PACK_PER_LAYER * 2);
    float*  stat   = (float*) carve(128 * 4);          // [sum(64) | sumsq(64)]
    float*  psum   = (float*) carve((size_t)G * (FDIM + 1) * 4);
    float*  pcnt   = psum + (size_t)G * FDIM;
    (void)ws_size; (void)n_in;

    const int ptotal = NCONV * 2 * KT * NT * 32;
    pack_weights<<<(ptotal + 255) / 256, 256, 0, stream>>>(lnf_W, lns_W, packed);

    embed_kernel<<<N, FDIM, 0, stream>>>(x, emb_W, emb_b, h);

    const long nf = (long)N * FDIM;
    const int nfBlocks = (int)((nf + 255) / 256);

    for (int i = 0; i < NCONV; ++i) {
        // BN1 stats
        zero_f32<<<1, 128, 0, stream>>>(stat, 128);
        bn_stats<<<128, 256, 0, stream>>>(h, stat, stat + 64, N);
        // BN1 apply + bf16 cast + zero agg
        bn_apply_cast<<<nfBlocks, 256, 0, stream>>>(h, stat, stat + 64,
                                                    bn1_g + i * FDIM, bn1_b + i * FDIM,
                                                    N, xn, xnb, agg);
        // fused gather + dual bf16 WMMA GEMM + gate + scatter
        edge_conv_wmma<<<1250, 256, 0, stream>>>(
            xnb, ei, ei + E, ea,
            packed + (size_t)i * PACK_PER_LAYER,
            lnf_b + i * FDIM, lns_b + i * FDIM,
            agg, ntiles);
        // BN2 stats on agg
        zero_f32<<<1, 128, 0, stream>>>(stat, 128);
        bn_stats<<<128, 256, 0, stream>>>(agg, stat, stat + 64, N);
        // BN2 apply + residual + softplus -> h
        bn2_residual<<<nfBlocks, 256, 0, stream>>>(agg, stat, stat + 64,
                                                   bn2_g + i * FDIM, bn2_b + i * FDIM,
                                                   N, xn, h);
    }

    // global mean pool + readout
    zero_f32<<<(G * (FDIM + 1) + 255) / 256, 256, 0, stream>>>(psum, (long)G * (FDIM + 1));
    pool_kernel<<<nfBlocks, 256, 0, stream>>>(h, batch, psum, pcnt, N);
    readout_kernel<<<G, HDIM, 0, stream>>>(psum, pcnt, fc1_W, fc1_b, out_W, out_b, out);
}
